// HeteroGNN_35974646071561
// MI455X (gfx1250) — matrix operand order repaired and checked by reading, not compiled, not run on hardware
//
#include <hip/hip_runtime.h>
#include <hip/hip_bf16.h>

#define NN 50000
#define EE 250000
#define GG 256
#define HH 64
#define HEADS 4
#define DH 16
#define LL 2

typedef __attribute__((ext_vector_type(16))) _Float16 v16h;
typedef __attribute__((ext_vector_type(8)))  float    v8f;

// ---------- helpers ----------
__device__ __forceinline__ unsigned fkey(float f) {
    unsigned u = __float_as_uint(f);
    return (u & 0x80000000u) ? ~u : (u | 0x80000000u);
}
__device__ __forceinline__ float funkey(unsigned u) {
    unsigned b = (u & 0x80000000u) ? (u & 0x7FFFFFFFu) : ~u;
    return __uint_as_float(b);
}

// ---------- generic zero ----------
__global__ void k_zero_u32(unsigned* __restrict__ p, int n) {
    int i = blockIdx.x * blockDim.x + threadIdx.x;
    if (i < n) p[i] = 0u;
}

// ---------- weight convert f32 -> f16 ----------
__global__ void k_cvt_f16(const float* __restrict__ src, _Float16* __restrict__ dst, int n) {
    int i = blockIdx.x * blockDim.x + threadIdx.x;
    if (i < n) dst[i] = (_Float16)src[i];
}

// ---------- batch histogram ----------
__global__ void k_count_batch(const int* __restrict__ bt, const int* __restrict__ ba,
                              const int* __restrict__ bv, int* __restrict__ counts) {
    int idx = blockIdx.x * blockDim.x + threadIdx.x;
    if (idx >= 3 * NN) return;
    int m = idx / NN, n = idx - m * NN;
    const int* bp = (m == 0) ? bt : (m == 1) ? ba : bv;
    atomicAdd(&counts[m * GG + bp[n]], 1);
}

// ---------- exclusive scan (256 bins x 3 modalities, one block) ----------
__global__ void k_scan_starts(const int* __restrict__ counts, int* __restrict__ starts) {
    __shared__ int buf[GG];
    int t = threadIdx.x;
    for (int m = 0; m < 3; m++) {
        int v = counts[m * GG + t];
        buf[t] = v;
        __syncthreads();
        for (int off = 1; off < GG; off <<= 1) {
            int x = (t >= off) ? buf[t - off] : 0;
            __syncthreads();
            buf[t] += x;
            __syncthreads();
        }
        starts[m * GG + t] = buf[t] - v;   // exclusive
        __syncthreads();
    }
}

// ---------- input projection + bias + positional encoding ----------
// block: 256 threads = 4 rows x 64 cols; grid.x = NN/4
__global__ void k_input_proj(const float* __restrict__ x, const float* __restrict__ W,
                             const float* __restrict__ b, const int* __restrict__ batch,
                             const int* __restrict__ starts, float* __restrict__ hout, int K) {
    __shared__ float xs[4 * 300];
    int row0 = blockIdx.x * 4;
    int tid = threadIdx.x;
    for (int i = tid; i < 4 * K; i += 256)
        xs[i] = x[(size_t)(row0 + i / K) * K + (i % K)];
    __syncthreads();
    int c = tid & 63, r = tid >> 6;
    int node = row0 + r;
    float acc = b[c];
    const float* xr = xs + r * K;
    for (int k = 0; k < K; k++) acc += xr[k] * W[k * HH + c];
    // positional encoding: pe[posn, c]
    int posn = node - starts[batch[node]];
    int kk = c >> 1;
    float div = __expf(-(float)(2 * kk) * 0.14391156509245135f);  // ln(10000)/64
    float ang = (float)posn * div;
    acc += (c & 1) ? __cosf(ang) : __sinf(ang);
    hout[(size_t)node * HH + c] = acc;
}

// ---------- node linear via WMMA: hp = h @ W(64x64) + b ----------
// one wave per 16-row tile; 4 col tiles x 2 K-chunks => 8 v_wmma per wave
__global__ void k_node_linear_wmma(const float* __restrict__ hin_base,
                                   const _Float16* __restrict__ Wh_base,
                                   const float* __restrict__ bias_base,
                                   float* __restrict__ hout_base) {
    int m = blockIdx.y;
    const float*    hin  = hin_base  + (size_t)m * NN * HH;
    const _Float16* Wh   = Wh_base   + (size_t)m * HH * HH;
    const float*    bias = bias_base + (size_t)m * HH;
    float*          hout = hout_base + (size_t)m * NN * HH;

    int wave = threadIdx.x >> 5;
    int lane = threadIdx.x & 31;
    int tile = blockIdx.x * (blockDim.x >> 5) + wave;
    if (tile >= NN / 16) return;   // wave-uniform exit, EXEC all-ones for WMMA
    int row0 = tile * 16;
    int mr = lane & 15, half = lane >> 4;

    // A fragments (16x32 f16 layout, ISA 7.12.2): two K chunks
    v16h a0, a1;
    const float* arow = hin + (size_t)(row0 + mr) * HH;
#pragma unroll
    for (int j = 0; j < 16; j++) {
        int vg = j >> 1, rb = j & 1;
        int kl = (vg < 4) ? (half * 8 + 2 * vg + rb)
                          : (16 + half * 8 + 2 * (vg - 4) + rb);
        a0[j] = (_Float16)arow[kl];
        a1[j] = (_Float16)arow[32 + kl];
    }
    int kb = lane & 15;   // B lane -> N column within tile
#pragma unroll
    for (int t = 0; t < 4; t++) {
        v16h b0, b1;
#pragma unroll
        for (int j = 0; j < 16; j++) {
            int kr = (j < 8) ? (half * 8 + j) : (16 + half * 8 + (j - 8));
            b0[j] = Wh[(size_t)kr * HH + t * 16 + kb];
            b1[j] = Wh[(size_t)(32 + kr) * HH + t * 16 + kb];
        }
        v8f c = {};
        c = __builtin_amdgcn_wmma_f32_16x16x32_f16(false, a0, false, b0, (short)0, c, false, false);
        c = __builtin_amdgcn_wmma_f32_16x16x32_f16(false, a1, false, b1, (short)0, c, false, false);
        int col = t * 16 + (lane & 15);
        float bv = bias[col];
#pragma unroll
        for (int vv = 0; vv < 8; vv++) {
            int rr = vv + 8 * half;     // C/D layout: VGPR v -> row v + 8*half
            hout[(size_t)(row0 + rr) * HH + col] = c[vv] + bv;
        }
    }
}

// ---------- GAT pass A: logits + segment max ----------
__global__ void k_gat_logits(const float* __restrict__ hp_src, const float* __restrict__ hp_dst,
                             const int* __restrict__ src, const int* __restrict__ dst,
                             const float* __restrict__ attv, float* __restrict__ logitbuf,
                             unsigned* __restrict__ smax) {
    int idx = blockIdx.x * blockDim.x + threadIdx.x;
    if (idx >= EE * HEADS) return;
    int e = idx >> 2, h = idx & 3;
    int s = src[e], d = dst[e];
    const float4* ps = (const float4*)(hp_src + (size_t)s * HH + h * DH);
    const float4* pd = (const float4*)(hp_dst + (size_t)d * HH + h * DH);
    const float4* pa = (const float4*)(attv + h * DH);
    float logit = 0.f;
#pragma unroll
    for (int q = 0; q < 4; q++) {
        float4 a = ps[q], b = pd[q], at = pa[q];
        float z;
        z = a.x + b.x; z = z > 0.f ? z : 0.2f * z; logit += z * at.x;
        z = a.y + b.y; z = z > 0.f ? z : 0.2f * z; logit += z * at.y;
        z = a.z + b.z; z = z > 0.f ? z : 0.2f * z; logit += z * at.z;
        z = a.w + b.w; z = z > 0.f ? z : 0.2f * z; logit += z * at.w;
    }
    logitbuf[idx] = logit;
    atomicMax(smax + (size_t)d * HEADS + h, fkey(logit));
}

// ---------- GAT pass B: exp(logit - max) + segment sum ----------
__global__ void k_gat_expsum(const int* __restrict__ dst, const unsigned* __restrict__ smax,
                             float* __restrict__ logitbuf, float* __restrict__ ssum) {
    int idx = blockIdx.x * blockDim.x + threadIdx.x;
    if (idx >= EE * HEADS) return;
    int e = idx >> 2, h = idx & 3;
    int d = dst[e];
    float mx = funkey(smax[(size_t)d * HEADS + h]);
    float a = __expf(logitbuf[idx] - mx);
    logitbuf[idx] = a;
    atomicAdd(&ssum[(size_t)d * HEADS + h], a);
}

// ---------- GAT pass C: acc[dst] += alpha * hs ----------
__global__ void k_gat_scatter(const float* __restrict__ hp_src, const int* __restrict__ src,
                              const int* __restrict__ dst, const float* __restrict__ logitbuf,
                              const float* __restrict__ ssum, float* __restrict__ acc) {
    int idx = blockIdx.x * blockDim.x + threadIdx.x;
    if (idx >= EE * HEADS) return;
    int e = idx >> 2, h = idx & 3;
    int s = src[e], d = dst[e];
    float alpha = logitbuf[idx] / ssum[(size_t)d * HEADS + h];
    const float4* ps = (const float4*)(hp_src + (size_t)s * HH + h * DH);
    float* pd = acc + (size_t)d * HH + h * DH;
#pragma unroll
    for (int q = 0; q < 4; q++) {
        float4 v = ps[q];
        atomicAdd(pd + q * 4 + 0, alpha * v.x);
        atomicAdd(pd + q * 4 + 1, alpha * v.y);
        atomicAdd(pd + q * 4 + 2, alpha * v.z);
        atomicAdd(pd + q * 4 + 3, alpha * v.w);
    }
}

// ---------- h = relu(acc / 3) ----------
__global__ void k_relu_div3(float* __restrict__ p, int n) {
    int i = blockIdx.x * blockDim.x + threadIdx.x;
    if (i < n) {
        float v = p[i] * (1.f / 3.f);
        p[i] = v > 0.f ? v : 0.f;
    }
}

// ---------- graph mean pool (atomic sum + count) ----------
__global__ void k_pool(const float* __restrict__ hA, const int* __restrict__ bt,
                       const int* __restrict__ ba, const int* __restrict__ bv,
                       float* __restrict__ pool, float* __restrict__ pcount) {
    int idx = blockIdx.x * blockDim.x + threadIdx.x;
    if (idx >= 3 * NN * 4) return;
    int node = idx >> 2, cg = idx & 3;
    int m = node / NN, n = node - m * NN;
    const int* bp = (m == 0) ? bt : (m == 1) ? ba : bv;
    int g = bp[n];
    const float4* src = (const float4*)(hA + (size_t)node * HH + cg * 16);
    float* dp = pool + (size_t)g * HH + cg * 16;
#pragma unroll
    for (int q = 0; q < 4; q++) {
        float4 v = src[q];
        atomicAdd(dp + q * 4 + 0, v.x);
        atomicAdd(dp + q * 4 + 1, v.y);
        atomicAdd(dp + q * 4 + 2, v.z);
        atomicAdd(dp + q * 4 + 3, v.w);
    }
    if (cg == 0) atomicAdd(&pcount[g], 1.0f);
}

// ---------- final MLP: 64 -> 16 -> 16 -> 1, one thread per graph ----------
__global__ void k_final_mlp(const float* __restrict__ pool, const float* __restrict__ pcount,
                            const float* __restrict__ Wf1, const float* __restrict__ bf1,
                            const float* __restrict__ Wf2, const float* __restrict__ bf2,
                            const float* __restrict__ Wf3, const float* __restrict__ bf3,
                            float* __restrict__ out) {
    int g = threadIdx.x;
    float cnt = pcount[g];
    cnt = cnt > 1.f ? cnt : 1.f;
    float inv = 1.f / cnt;
    float y1[16];
#pragma unroll
    for (int o = 0; o < 16; o++) y1[o] = bf1[o];
    for (int k = 0; k < HH; k++) {
        float gv = pool[(size_t)g * HH + k] * inv;
#pragma unroll
        for (int o = 0; o < 16; o++) y1[o] += gv * Wf1[k * 16 + o];
    }
    float y2[16];
#pragma unroll
    for (int o = 0; o < 16; o++) { y1[o] = y1[o] > 0.f ? y1[o] : 0.f; y2[o] = bf2[o]; }
    for (int k = 0; k < 16; k++)
#pragma unroll
        for (int o = 0; o < 16; o++) y2[o] += y1[k] * Wf2[k * 16 + o];
    float y3 = bf3[0];
#pragma unroll
    for (int k = 0; k < 16; k++) {
        float t = y2[k] > 0.f ? y2[k] : 0.f;
        y3 += t * Wf3[k];
    }
    out[g] = y3;
}

extern "C" void kernel_launch(void* const* d_in, const int* in_sizes, int n_in,
                              void* d_out, int out_size, void* d_ws, size_t ws_size,
                              hipStream_t stream) {
    (void)in_sizes; (void)n_in; (void)out_size; (void)ws_size;
    const float* x_text  = (const float*)d_in[0];
    const float* x_audio = (const float*)d_in[1];
    const float* x_video = (const float*)d_in[2];
    const int*   batch_t = (const int*)d_in[3];
    const int*   batch_a = (const int*)d_in[4];
    const int*   batch_v = (const int*)d_in[5];
    const int*   edges   = (const int*)d_in[6];
    const float* W_in_t  = (const float*)d_in[7];
    const float* W_in_a  = (const float*)d_in[8];
    const float* W_in_v  = (const float*)d_in[9];
    const float* b_in_t  = (const float*)d_in[10];
    const float* b_in_a  = (const float*)d_in[11];
    const float* b_in_v  = (const float*)d_in[12];
    const float* W_node  = (const float*)d_in[13];
    const float* b_node  = (const float*)d_in[14];
    const float* att     = (const float*)d_in[15];
    const float* Wf1 = (const float*)d_in[16];
    const float* bf1 = (const float*)d_in[17];
    const float* Wf2 = (const float*)d_in[18];
    const float* bf2 = (const float*)d_in[19];
    const float* Wf3 = (const float*)d_in[20];
    const float* bf3 = (const float*)d_in[21];
    float* out = (float*)d_out;

    // workspace carve-up (~83 MB total)
    char* ws = (char*)d_ws;
    size_t off = 0;
    auto take = [&](size_t bytes) -> char* {
        char* p = ws + off;
        off = (off + bytes + 255) & ~(size_t)255;
        return p;
    };
    float*     hA       = (float*)take((size_t)3 * NN * HH * 4);  // h / acc (double duty)
    float*     hB       = (float*)take((size_t)3 * NN * HH * 4);  // hp
    float*     logitbuf = (float*)take((size_t)EE * HEADS * 4);
    unsigned*  smax     = (unsigned*)take((size_t)NN * HEADS * 4);
    float*     ssum     = (float*)take((size_t)NN * HEADS * 4);
    int*       counts   = (int*)take((size_t)3 * GG * 4);
    int*       starts   = (int*)take((size_t)3 * GG * 4);
    _Float16*  Wh16     = (_Float16*)take((size_t)LL * 3 * HH * HH * 2);
    float*     pool     = (float*)take((size_t)GG * HH * 4);
    float*     pcount   = (float*)take((size_t)GG * 4);

    const int B = 256;
    auto nb = [](long n) { return (int)((n + 255) / 256); };

    // 0) convert node-linear weights to f16 (once per call; deterministic)
    k_cvt_f16<<<nb(LL * 3 * HH * HH), B, 0, stream>>>(W_node, Wh16, LL * 3 * HH * HH);

    // 1) segment starts for positional encoding
    k_zero_u32<<<nb(3 * GG), B, 0, stream>>>((unsigned*)counts, 3 * GG);
    k_count_batch<<<nb(3L * NN), B, 0, stream>>>(batch_t, batch_a, batch_v, counts);
    k_scan_starts<<<1, GG, 0, stream>>>(counts, starts);

    // 2) input projections + bias + positional encoding -> hA
    k_input_proj<<<NN / 4, B, 0, stream>>>(x_text,  W_in_t, b_in_t, batch_t, starts + 0 * GG, hA + (size_t)0 * NN * HH, 300);
    k_input_proj<<<NN / 4, B, 0, stream>>>(x_audio, W_in_a, b_in_a, batch_a, starts + 1 * GG, hA + (size_t)1 * NN * HH, 74);
    k_input_proj<<<NN / 4, B, 0, stream>>>(x_video, W_in_v, b_in_v, batch_v, starts + 2 * GG, hA + (size_t)2 * NN * HH, 35);

    // 3) GAT layers
    const int tiles = NN / 16;                    // 3125
    const int wpb = B / 32;                       // 8 waves per block
    dim3 wgrid((tiles + wpb - 1) / wpb, 3);
    for (int l = 0; l < LL; l++) {
        // hp = h @ W + b   (WMMA, per modality via grid.y)
        k_node_linear_wmma<<<wgrid, B, 0, stream>>>(
            hA, Wh16 + (size_t)l * 3 * HH * HH, b_node + (size_t)l * 3 * HH, hB);
        // acc (= hA) <- 0
        k_zero_u32<<<nb(3L * NN * HH), B, 0, stream>>>((unsigned*)hA, 3 * NN * HH);
        for (int e_id = 0; e_id < 9; e_id++) {
            int i = e_id / 3, j = e_id % 3;
            const int* src = edges + (size_t)e_id * 2 * EE;
            const int* dst = src + EE;
            const float* hp_src = hB + (size_t)i * NN * HH;
            const float* hp_dst = hB + (size_t)j * NN * HH;
            const float* attv = att + ((size_t)l * 9 + e_id) * HEADS * DH;
            k_zero_u32<<<nb((long)NN * HEADS), B, 0, stream>>>(smax, NN * HEADS);
            k_zero_u32<<<nb((long)NN * HEADS), B, 0, stream>>>((unsigned*)ssum, NN * HEADS);
            k_gat_logits <<<nb((long)EE * HEADS), B, 0, stream>>>(hp_src, hp_dst, src, dst, attv, logitbuf, smax);
            k_gat_expsum <<<nb((long)EE * HEADS), B, 0, stream>>>(dst, smax, logitbuf, ssum);
            k_gat_scatter<<<nb((long)EE * HEADS), B, 0, stream>>>(hp_src, src, dst, logitbuf, ssum,
                                                                  hA + (size_t)j * NN * HH);
        }
        // h = relu(acc / 3)
        k_relu_div3<<<nb(3L * NN * HH), B, 0, stream>>>(hA, 3 * NN * HH);
    }

    // 4) mean pooling over graphs
    k_zero_u32<<<nb(GG * HH), B, 0, stream>>>((unsigned*)pool, GG * HH);
    k_zero_u32<<<nb(GG), B, 0, stream>>>((unsigned*)pcount, GG);
    k_pool<<<nb(3L * NN * 4), B, 0, stream>>>(hA, batch_t, batch_a, batch_v, pool, pcount);

    // 5) final MLP -> out (G,)
    k_final_mlp<<<1, GG, 0, stream>>>(pool, pcount, Wf1, bf1, Wf2, bf2, Wf3, bf3, out);
}
